// ContrastiveLoss_47931835023666
// MI455X (gfx1250) — matrix operand order repaired and checked by reading, not compiled, not run on hardware
//
#include <hip/hip_runtime.h>
#include <hip/hip_bf16.h>

typedef __attribute__((ext_vector_type(16))) _Float16 v16h;
typedef __attribute__((ext_vector_type(8)))  float    v8f;

#define CN 8192   // batch size
#define CD 256    // embedding dim

union FragU {
    v16h  h;
    uint4 q[2];
};

// ---------------------------------------------------------------------------
// Kernel 1: L2-normalize E and Q rows -> f16 arrays; zero rowsum accumulator.
// One 256-thread block per row (tid == column index, D == 256).
// ---------------------------------------------------------------------------
__global__ __launch_bounds__(256)
void cl_normalize_kernel(const float* __restrict__ E, const float* __restrict__ Q,
                         _Float16* __restrict__ Ah, _Float16* __restrict__ Bh,
                         float* __restrict__ rowsum) {
    const int row = blockIdx.x;
    const int tid = threadIdx.x;
    const float e = E[(size_t)row * CD + tid];
    const float q = Q[(size_t)row * CD + tid];
    float se = e * e, sq = q * q;
    // wave32 reduction
    for (int m = 16; m >= 1; m >>= 1) {
        se += __shfl_xor(se, m, 32);
        sq += __shfl_xor(sq, m, 32);
    }
    __shared__ float shE[8], shQ[8];
    const int wave = tid >> 5, lane = tid & 31;
    if (lane == 0) { shE[wave] = se; shQ[wave] = sq; }
    __syncthreads();
    se = 0.f; sq = 0.f;
    #pragma unroll
    for (int w = 0; w < 8; ++w) { se += shE[w]; sq += shQ[w]; }
    const float re = 1.0f / fmaxf(sqrtf(se), 1e-8f);  // x / max(||x||, eps)
    const float rq = 1.0f / fmaxf(sqrtf(sq), 1e-8f);
    Ah[(size_t)row * CD + tid] = (_Float16)(e * re);
    Bh[(size_t)row * CD + tid] = (_Float16)(q * rq);
    if (tid == 0) rowsum[row] = 0.0f;
}

// ---------------------------------------------------------------------------
// Kernel 2: per-row diagonal sim[i,i] and neighbor sim[i, nbr(i)] terms.
// One wave32 per row, 8 rows per block.
// ---------------------------------------------------------------------------
__global__ __launch_bounds__(256)
void cl_diag_kernel(const _Float16* __restrict__ Ah, const _Float16* __restrict__ Bh,
                    float* __restrict__ diag, float* __restrict__ nbr) {
    const int wave = threadIdx.x >> 5;
    const int lane = threadIdx.x & 31;
    const int i = blockIdx.x * 8 + wave;
    const int j = (i < CN - 1) ? (i + 1) : (CN - 2);  // clamped target column
    float d = 0.f, nb = 0.f;
    #pragma unroll
    for (int k = lane; k < CD; k += 32) {
        const float a = (float)Ah[(size_t)i * CD + k];
        d  += a * (float)Bh[(size_t)i * CD + k];
        nb += a * (float)Bh[(size_t)j * CD + k];
    }
    for (int m = 16; m >= 1; m >>= 1) {
        d  += __shfl_xor(d,  m, 32);
        nb += __shfl_xor(nb, m, 32);
    }
    if (lane == 0) { diag[i] = d; nbr[i] = nb; }
}

// ---------------------------------------------------------------------------
// Kernel 3: WMMA GEMM sim = Ah @ Bh^T fused with exp(-sim-1) row reduction.
// 128x128 tile per block; 8 waves in a 4(M) x 2(N) grid; each wave computes
// a 32x64 sub-tile as 2x4 v_wmma_f32_16x16x32_f16 accumulators.
// rowsum[i] += sum_j exp(-sim[i,j] - 1)   (includes j==i; fixed shift is safe
// because |sim| <= 1, so all exp args lie in [-2, 0]).
// ---------------------------------------------------------------------------
__global__ __launch_bounds__(256)
void cl_simexp_kernel(const _Float16* __restrict__ Ah, const _Float16* __restrict__ Bh,
                      float* __restrict__ rowsum) {
    const int tid  = threadIdx.x;
    const int wave = tid >> 5;
    const int lane = tid & 31;
    const int hlf  = lane >> 4;     // half-wave select (ISA fragment layouts)
    const int l16  = lane & 15;
    const int wm   = wave & 3;      // wave M coordinate (0..3)
    const int wn   = wave >> 2;     // wave N coordinate (0..1)
    const int row0 = blockIdx.x * 128 + wm * 32;
    const int col0 = blockIdx.y * 128 + wn * 64;

    v8f acc[2][4] = {};

    for (int k0 = 0; k0 < CD; k0 += 32) {
        FragU a[2], b[4];
        // A fragment, 16-bit 16x32 layout: lane l16 -> row M, element e -> K =
        // (e/8)*16 + hlf*8 + (e%8): two contiguous 16-byte chunks per lane.
        #pragma unroll
        for (int mi = 0; mi < 2; ++mi) {
            const _Float16* p = Ah + (size_t)(row0 + mi * 16 + l16) * CD + k0 + hlf * 8;
            a[mi].q[0] = *(const uint4*)(p);
            a[mi].q[1] = *(const uint4*)(p + 16);
        }
        // B fragment, 16-bit 32x16 layout: lane l16 -> column N, element e ->
        // K = (e/8)*16 + hlf*8 + (e%8): two contiguous 16-byte chunks per lane.
        #pragma unroll
        for (int ni = 0; ni < 4; ++ni) {
            const _Float16* p = Bh + (size_t)(col0 + ni * 16 + l16) * CD + k0 + hlf * 8;
            b[ni].q[0] = *(const uint4*)(p);
            b[ni].q[1] = *(const uint4*)(p + 16);
        }
        #pragma unroll
        for (int mi = 0; mi < 2; ++mi)
            #pragma unroll
            for (int ni = 0; ni < 4; ++ni)
                acc[mi][ni] = __builtin_amdgcn_wmma_f32_16x16x32_f16(
                    false, a[mi].h, false, b[ni].h,
                    (short)0, acc[mi][ni], false, false);
    }

    // C/D layout: VGPR r holds M = r + 8*hlf, lane l16 holds N = l16.
    #pragma unroll
    for (int mi = 0; mi < 2; ++mi) {
        float rs[8];
        #pragma unroll
        for (int r = 0; r < 8; ++r) {
            float s = 0.f;
            #pragma unroll
            for (int ni = 0; ni < 4; ++ni)
                s += __expf(-acc[mi][ni][r] - 1.0f);
            rs[r] = s;
        }
        // reduce across the 16 columns (lane bits 0..3; bit 4 = row-half stays put)
        #pragma unroll
        for (int r = 0; r < 8; ++r)
            for (int m = 1; m <= 8; m <<= 1)
                rs[r] += __shfl_xor(rs[r], m, 32);
        if (l16 == 0) {
            #pragma unroll
            for (int r = 0; r < 8; ++r)
                atomicAdd(rowsum + row0 + mi * 16 + hlf * 8 + r, rs[r]);
        }
    }
}

// ---------------------------------------------------------------------------
// Kernel 4: final reduction.
// loss = mean_i [ 1 + log(rowsum[i] - exp(-diag[i]-1)) + nbr[i] ]   (+ loss_pos==0)
// ---------------------------------------------------------------------------
__global__ __launch_bounds__(256)
void cl_finish_kernel(const float* __restrict__ rowsum, const float* __restrict__ diag,
                      const float* __restrict__ nbr, float* __restrict__ out) {
    __shared__ float sh[256];
    const int tid = threadIdx.x;
    float acc = 0.f;
    for (int i = tid; i < CN; i += 256) {
        const float s = rowsum[i] - __expf(-diag[i] - 1.0f);  // drop diagonal term
        acc += logf(s) + 1.0f + nbr[i];
    }
    sh[tid] = acc;
    __syncthreads();
    for (int s = 128; s > 0; s >>= 1) {
        if (tid < s) sh[tid] += sh[tid + s];
        __syncthreads();
    }
    if (tid == 0) out[0] = sh[0] / (float)CN;
}

extern "C" void kernel_launch(void* const* d_in, const int* in_sizes, int n_in,
                              void* d_out, int out_size, void* d_ws, size_t ws_size,
                              hipStream_t stream) {
    const float* E = (const float*)d_in[0];   // embeddings        (N, D) f32
    const float* Q = (const float*)d_in[1];   // query_embeddings  (N, D) f32

    char* ws = (char*)d_ws;
    _Float16* Ah     = (_Float16*)(ws);                                   // 4 MB
    _Float16* Bh     = (_Float16*)(ws + (size_t)CN * CD * sizeof(_Float16));
    float*    rowsum = (float*)   (ws + (size_t)CN * CD * sizeof(_Float16) * 2);
    float*    diag   = rowsum + CN;
    float*    nbr    = diag + CN;

    cl_normalize_kernel<<<CN, 256, 0, stream>>>(E, Q, Ah, Bh, rowsum);
    cl_diag_kernel<<<CN / 8, 256, 0, stream>>>(Ah, Bh, diag, nbr);

    dim3 grid(CN / 128, CN / 128);
    cl_simexp_kernel<<<grid, 256, 0, stream>>>(Ah, Bh, rowsum);

    cl_finish_kernel<<<1, 256, 0, stream>>>(rowsum, diag, nbr, (float*)d_out);
}
